// ExclusiveSelfAttention_3289944949278
// MI455X (gfx1250) — matrix operand order repaired and checked by reading, not compiled, not run on hardware
//
#include <hip/hip_runtime.h>
#include <hip/hip_bf16.h>
#include <cstdint>
#include <cstddef>

typedef unsigned short u16;
typedef __bf16 bf16_t;
typedef bf16_t v16bf __attribute__((ext_vector_type(16)));
typedef float  v8f   __attribute__((ext_vector_type(8)));

union Frag {
    v16bf v;
    u16   h[16];
    uint4 q[2];
};

__device__ __forceinline__ u16 f2bf(float f) {
    union { float f; unsigned u; } x; x.f = f;
    unsigned r = x.u + 0x7FFFu + ((x.u >> 16) & 1u);   // round-to-nearest-even
    return (u16)(r >> 16);
}
__device__ __forceinline__ float bf2f(u16 h) {
    union { unsigned u; float f; } x; x.u = ((unsigned)h) << 16;
    return x.f;
}

constexpr int DM = 1024;   // d_model
constexpr int HD = 64;     // head dim
constexpr int NH = 16;     // heads

// ---------------------------------------------------------------- convert f32 -> bf16
__global__ __launch_bounds__(256) void cvt_bf16_kernel(const float* __restrict__ s,
                                                       u16* __restrict__ d, int n) {
    int i = blockIdx.x * 256 + threadIdx.x;
    if (i < n) d[i] = f2bf(s[i]);
}

// ---------------------------------------------------------------- GEMM: Y = X @ W^T
// X: M x K bf16 (row major), W: N x K bf16 (row major), Y: M x N
// block = 128 threads (4 waves, 2x2); wave tile 64(M) x 64(N); block tile 128 x 128.
// Per K-step: 8 A/B fragments (16 b128 loads) feed 16 WMMAs -> 4x register reuse each way.
__global__ __launch_bounds__(128) void gemm_xwt_bf16(const u16* __restrict__ X,
                                                     const u16* __restrict__ W,
                                                     float* __restrict__ Yf,
                                                     u16*  __restrict__ Yh,
                                                     int M, int N, int K) {
    const int lane = threadIdx.x & 31;
    const int wave = threadIdx.x >> 5;
    const int hf   = lane >> 4;      // lane half (0: lanes 0-15, 1: lanes 16-31)
    const int lr   = lane & 15;
    const int m0 = blockIdx.y * 128 + (wave >> 1) * 64;
    const int n0 = blockIdx.x * 128 + (wave & 1)  * 64;

    v8f acc[4][4];
    for (int i = 0; i < 4; ++i)
        for (int j = 0; j < 4; ++j)
            for (int e = 0; e < 8; ++e) acc[i][j][e] = 0.0f;

    for (int k0 = 0; k0 < K; k0 += 32) {
        // A fragments (16x32 bf16): lane<16 -> K {0..7,16..23}; lane>=16 -> {8..15,24..31}
        Frag a[4];
#pragma unroll
        for (int im = 0; im < 4; ++im) {
            const int row = m0 + im * 16 + lr;
            const uint4* p = (const uint4*)(X + (size_t)row * K + k0 + hf * 8);
            a[im].q[0] = p[0];     // K: k0+hf*8 .. +7
            a[im].q[1] = p[2];     // K: k0+16+hf*8 .. +7
        }
        // B fragments (32x16 bf16): lane<16 -> col n0', K {0..7,16..23}; lane>=16 -> {8..15,24..31}
        Frag b[4];
#pragma unroll
        for (int in = 0; in < 4; ++in) {
            const int col = n0 + in * 16 + lr;
            const uint4* p = (const uint4*)(W + (size_t)col * K + k0 + hf * 8);
            b[in].q[0] = p[0];
            b[in].q[1] = p[2];
        }
#pragma unroll
        for (int im = 0; im < 4; ++im)
#pragma unroll
            for (int in = 0; in < 4; ++in)
                acc[im][in] = __builtin_amdgcn_wmma_f32_16x16x32_bf16(
                    false, a[im].v, false, b[in].v, (short)0, acc[im][in], false, false);
    }

#pragma unroll
    for (int im = 0; im < 4; ++im)
#pragma unroll
        for (int in = 0; in < 4; ++in)
#pragma unroll
            for (int j = 0; j < 8; ++j) {
                const int row = m0 + im * 16 + j + 8 * hf;   // C layout: M = vgpr + 8*half
                const int col = n0 + in * 16 + lr;           //           N = lane%16
                const float vv = acc[im][in][j];
                if (Yf) Yf[(size_t)row * N + col] = vv;
                if (Yh) Yh[(size_t)row * N + col] = f2bf(vv);
            }
}

// ---------------------------------------------------------------- flash attention
// One block = (b, h, 64-query tile); 4 waves, each owns 16 query rows.
// K tile is staged global->LDS with async-to-LDS copies (ASYNCcnt path);
// V tile is transposed through VGPRs so its B-fragments are contiguous LDS reads.
__global__ __launch_bounds__(128) void attn_kernel(const u16* __restrict__ Q,
                                                   const u16* __restrict__ Kb,
                                                   const u16* __restrict__ Vb,
                                                   float* __restrict__ O, int S) {
    __shared__ __align__(16) u16 Klds[32 * 64];   // [kv][d]
    __shared__ __align__(16) u16 Vt[64 * 32];     // [d][kv]  (transposed for B-frag reads)
    __shared__ __align__(16) u16 Pst[4 * 16 * 32];

    const int lane = threadIdx.x & 31;
    const int wave = threadIdx.x >> 5;
    const int hf   = lane >> 4;
    const int lr   = lane & 15;

    const int qt = blockIdx.x, h = blockIdx.y, b = blockIdx.z;
    const int q0 = qt * 64 + wave * 16;

    const size_t headoff = (size_t)h * HD;
    const size_t basebs  = (size_t)b * S;

    // Q A-fragments for the two 32-wide d chunks
    Frag qf[2];
#pragma unroll
    for (int c = 0; c < 2; ++c) {
        const uint4* p = (const uint4*)(Q + (basebs + q0 + lr) * DM + headoff + 32 * c + hf * 8);
        qf[c].q[0] = p[0];
        qf[c].q[1] = p[2];
    }

    v8f o[4];
    for (int n = 0; n < 4; ++n)
        for (int e = 0; e < 8; ++e) o[n][e] = 0.0f;
    float mrow[8], lrow[8];
    for (int j = 0; j < 8; ++j) { mrow[j] = -3.0e38f; lrow[j] = 0.0f; }

    const int tks = threadIdx.x >> 2;          // kv row in tile 0..31
    const int tds = (threadIdx.x & 3) * 16;    // d start
    const float scale = 0.125f;                // 1/sqrt(64)

    // per-thread LDS destination (byte address) for the async K-tile copy
    const unsigned kldsa = (unsigned)(uintptr_t)(Klds + tks * 64 + tds);

    for (int kv0 = 0; kv0 < S; kv0 += 32) {
        // ---- stage K tile with async global->LDS copies (no VGPR round-trip)
        {
            const u16* gk = Kb + (basebs + kv0 + tks) * DM + headoff + tds;
            asm volatile(
                "global_load_async_to_lds_b128 %0, %1, off\n\t"
                "global_load_async_to_lds_b128 %0, %1, off offset:16"
                :: "v"(kldsa), "v"(gk) : "memory");
        }
        // ---- stage V tile transposed through VGPRs
        {
            Frag tv;
            const uint4* gv = (const uint4*)(Vb + (basebs + kv0 + tks) * DM + headoff + tds);
            tv.q[0] = gv[0]; tv.q[1] = gv[1];
#pragma unroll
            for (int e = 0; e < 16; ++e) Vt[(tds + e) * 32 + tks] = tv.h[e];
        }
        asm volatile("s_wait_asynccnt 0x0" ::: "memory");
        __syncthreads();

        // scores S = Q @ K^T  (two 16x16 C tiles; K-dim = 64 -> 2 WMMA each)
        v8f s[2];
#pragma unroll
        for (int t = 0; t < 2; ++t) {
            v8f accs; for (int e = 0; e < 8; ++e) accs[e] = 0.0f;
#pragma unroll
            for (int c = 0; c < 2; ++c) {
                Frag kf;
                const uint4* p = (const uint4*)(Klds + (16 * t + lr) * 64 + 32 * c + hf * 8);
                kf.q[0] = p[0]; kf.q[1] = p[2];
                accs = __builtin_amdgcn_wmma_f32_16x16x32_bf16(
                    false, qf[c].v, false, kf.v, (short)0, accs, false, false);
            }
            s[t] = accs;
        }

        // online softmax (row j lives across one 16-lane half -> xor masks 1,2,4,8)
        float alpha[8];
#pragma unroll
        for (int j = 0; j < 8; ++j) {
            float s0 = s[0][j] * scale, s1 = s[1][j] * scale;
            float rm = fmaxf(s0, s1);
            rm = fmaxf(rm, __shfl_xor(rm, 1, 32));
            rm = fmaxf(rm, __shfl_xor(rm, 2, 32));
            rm = fmaxf(rm, __shfl_xor(rm, 4, 32));
            rm = fmaxf(rm, __shfl_xor(rm, 8, 32));
            const float mnew = fmaxf(mrow[j], rm);
            alpha[j] = __expf(mrow[j] - mnew);
            mrow[j] = mnew;
            const float p0 = __expf(s0 - mnew);
            const float p1 = __expf(s1 - mnew);
            s[0][j] = p0; s[1][j] = p1;
            float rs = p0 + p1;
            rs += __shfl_xor(rs, 1, 32);
            rs += __shfl_xor(rs, 2, 32);
            rs += __shfl_xor(rs, 4, 32);
            rs += __shfl_xor(rs, 8, 32);
            lrow[j] = lrow[j] * alpha[j] + rs;
        }
#pragma unroll
        for (int n = 0; n < 4; ++n)
#pragma unroll
            for (int j = 0; j < 8; ++j) o[n][j] *= alpha[j];

        // bounce P through per-wave LDS: C layout -> A layout (bf16)
        u16* pp = Pst + wave * 512;
#pragma unroll
        for (int t = 0; t < 2; ++t)
#pragma unroll
            for (int j = 0; j < 8; ++j)
                pp[(j + 8 * hf) * 32 + 16 * t + lr] = f2bf(s[t][j]);

        Frag pa;
        {
            const uint4* p = (const uint4*)(pp + lr * 32 + hf * 8);
            pa.q[0] = p[0];
            pa.q[1] = p[2];
        }

        // o += P @ V   (V B-frags contiguous from transposed LDS tile)
#pragma unroll
        for (int n = 0; n < 4; ++n) {
            Frag vf;
            const uint4* p = (const uint4*)(Vt + (16 * n + lr) * 32 + hf * 8);
            vf.q[0] = p[0]; vf.q[1] = p[2];
            o[n] = __builtin_amdgcn_wmma_f32_16x16x32_bf16(
                false, pa.v, false, vf.v, (short)0, o[n], false, false);
        }
        __syncthreads();
    }

    // epilogue: divide by softmax denominator, write f32 (B,S,D) layout
#pragma unroll
    for (int n = 0; n < 4; ++n)
#pragma unroll
        for (int j = 0; j < 8; ++j) {
            const int row = q0 + j + 8 * hf;
            const int d   = 16 * n + lr;
            O[(basebs + row) * DM + headoff + d] = o[n][j] / lrow[j];
        }
}

// ---------------------------------------------------------------- exclusive projection fixup
// out -= xsa * (dot(out,v)/(|v|^2+eps)) * v ; one wave per (token, head)
__global__ __launch_bounds__(256) void fixup_kernel(const float* __restrict__ O,
                                                    const u16* __restrict__ Vb,
                                                    const float* __restrict__ xsa,
                                                    u16* __restrict__ Ob) {
    const int thid = blockIdx.x * 8 + (threadIdx.x >> 5);
    const int lane = threadIdx.x & 31;
    const int bs = thid >> 4;
    const int h  = thid & 15;
    const size_t base = (size_t)bs * DM + h * HD + lane * 2;
    const float o0 = O[base], o1 = O[base + 1];
    const float v0 = bf2f(Vb[base]), v1 = bf2f(Vb[base + 1]);
    float dot = o0 * v0 + o1 * v1;
    float nrm = v0 * v0 + v1 * v1;
#pragma unroll
    for (int m = 16; m >= 1; m >>= 1) {
        dot += __shfl_xor(dot, m, 32);
        nrm += __shfl_xor(nrm, m, 32);
    }
    const float coef = xsa[0] * dot / (nrm + 1e-8f);
    Ob[base]     = f2bf(o0 - coef * v0);
    Ob[base + 1] = f2bf(o1 - coef * v1);
}

// ---------------------------------------------------------------- host launcher
extern "C" void kernel_launch(void* const* d_in, const int* in_sizes, int n_in,
                              void* d_out, int out_size, void* d_ws, size_t ws_size,
                              hipStream_t stream) {
    (void)in_sizes; (void)n_in; (void)out_size; (void)ws_size;
    const int B = 2, S = 2048;
    const int M = B * S;  // 4096

    const float* x   = (const float*)d_in[0];
    const float* Wq  = (const float*)d_in[1];
    const float* Wk  = (const float*)d_in[2];
    const float* Wv  = (const float*)d_in[3];
    const float* Wo  = (const float*)d_in[4];
    const float* xsa = (const float*)d_in[5];

    char* ws = (char*)d_ws;
    const size_t MB = 1024u * 1024u;
    u16*   xb   = (u16*)(ws + 0 * MB);     // 8 MB
    u16*   wqb  = (u16*)(ws + 8 * MB);     // 2 MB each
    u16*   wkb  = (u16*)(ws + 10 * MB);
    u16*   wvb  = (u16*)(ws + 12 * MB);
    u16*   wob  = (u16*)(ws + 14 * MB);
    u16*   Qb   = (u16*)(ws + 16 * MB);    // 8 MB each
    u16*   Kb   = (u16*)(ws + 24 * MB);
    u16*   Vb   = (u16*)(ws + 32 * MB);
    float* Oatt = (float*)(ws + 40 * MB);  // 16 MB
    u16*   Ob   = (u16*)(ws + 56 * MB);    // 8 MB

    cvt_bf16_kernel<<<(M * DM) / 256, 256, 0, stream>>>(x, xb, M * DM);
    cvt_bf16_kernel<<<(DM * DM) / 256, 256, 0, stream>>>(Wq, wqb, DM * DM);
    cvt_bf16_kernel<<<(DM * DM) / 256, 256, 0, stream>>>(Wk, wkb, DM * DM);
    cvt_bf16_kernel<<<(DM * DM) / 256, 256, 0, stream>>>(Wv, wvb, DM * DM);
    cvt_bf16_kernel<<<(DM * DM) / 256, 256, 0, stream>>>(Wo, wob, DM * DM);

    dim3 ggrid(DM / 128, M / 128);  // (8, 32)
    gemm_xwt_bf16<<<ggrid, 128, 0, stream>>>(xb, wqb, nullptr, Qb, M, DM, DM);
    gemm_xwt_bf16<<<ggrid, 128, 0, stream>>>(xb, wkb, nullptr, Kb, M, DM, DM);
    gemm_xwt_bf16<<<ggrid, 128, 0, stream>>>(xb, wvb, nullptr, Vb, M, DM, DM);

    attn_kernel<<<dim3(S / 64, NH, B), 128, 0, stream>>>(Qb, Kb, Vb, Oatt, S);

    fixup_kernel<<<(B * S * NH) / 8, 256, 0, stream>>>(Oatt, Vb, xsa, Ob);

    gemm_xwt_bf16<<<ggrid, 128, 0, stream>>>(Ob, wob, (float*)d_out, nullptr, M, DM, DM);
}
